// RecurrentNet_35519379537968
// MI455X (gfx1250) — matrix operand order repaired and checked by reading, not compiled, not run on hardware
//
#include <hip/hip_runtime.h>

// ---------------------------------------------------------------------------
// Fused 2-layer Elman RNN (relu) for MI455X / gfx1250.
//   h0[t] = relu(x[t] @ Wih0^T + h0[t-1] @ Whh0^T + (bih0+bhh0))
//   h1[t] = relu(h0[t] @ Wih1^T + h1[t-1] @ Whh1^T + (bih1+bhh1))
//   out[b] = h1[b, T-1, H-1]
//
// Latency-bound sequential scan. 4 WGs (16 batch rows each, batch rows are
// independent recurrences), 8 waves/WG, each wave owns a 16-wide H slice.
// Weights live in VGPRs as pre-swizzled v_wmma_f32_16x16x32_bf16 B-operands
// for the whole kernel. Hidden state is exchanged through ping-pong LDS
// buffers (2 barriers/step instead of 4). x[t+1] operand fragments are
// software-pipelined in registers so global latency is off the critical path.
// ---------------------------------------------------------------------------

typedef __attribute__((ext_vector_type(16))) __bf16 v16bf;
typedef __attribute__((ext_vector_type(8)))  float  v8f;

#define B_TOT 64
#define T_LEN 2048
#define D_IN  64
#define H_DIM 128
#define LDS_STRIDE 136   // 128 + 8 bf16 pad -> 272B row stride, skews banks

// A-operand (16x32 bf16, MxK) fragment for one lane:
//   lanes 0-15  (M = lane):    halves 0..7 -> K0..K0+7, 8..15 -> K0+16..K0+23
//   lanes 16-31 (M = lane-16): same with K0 += 8        (K0 base = 32*kb)
// Both runs are 8 contiguous bf16 (16B) -> two ds_load_b128 per fragment.
static __device__ __forceinline__ v16bf loadA_lds(const __bf16* __restrict__ row, int K0) {
  v16bf a;
#pragma unroll
  for (int i = 0; i < 8; ++i) { a[i] = row[K0 + i]; a[8 + i] = row[K0 + 16 + i]; }
  return a;
}

static __device__ __forceinline__ v16bf loadA_f32(const float* __restrict__ row, int K0) {
  v16bf a;
#pragma unroll
  for (int i = 0; i < 8; ++i) {
    a[i]     = (__bf16)row[K0 + i];
    a[8 + i] = (__bf16)row[K0 + 16 + i];
  }
  return a;
}

// B-operand (32x16 bf16, KxN): lane n = lane&15 holds column n; elements 0..7
// -> K0..K0+7, 8..15 -> K0+16..K0+23; lanes 16-31 with K0 += 8 (matches A).
// two runs of 8 contiguous f32 from W row j (W is [H_out][K] row-major; B[k][n] = W[j][k]).
static __device__ __forceinline__ v16bf loadB_f32(const float* __restrict__ wrow, int K0) {
  v16bf b;
#pragma unroll
  for (int i = 0; i < 8; ++i) { b[i] = (__bf16)wrow[K0 + i]; b[8 + i] = (__bf16)wrow[K0 + 16 + i]; }
  return b;
}

static __device__ __forceinline__ float relu1(float v) { return v > 0.f ? v : 0.f; }

__global__ __launch_bounds__(256, 1)
void rnn2_fused(const float* __restrict__ x,
                const float* __restrict__ Wih0, const float* __restrict__ Whh0,
                const float* __restrict__ bih0, const float* __restrict__ bhh0,
                const float* __restrict__ Wih1, const float* __restrict__ Whh1,
                const float* __restrict__ bih1, const float* __restrict__ bhh1,
                float* __restrict__ out)
{
  __shared__ __align__(16) __bf16 h0s[2][16 * LDS_STRIDE];  // ping-pong
  __shared__ __align__(16) __bf16 h1s[2][16 * LDS_STRIDE];  // ping-pong

  const int  tid    = threadIdx.x;
  const int  wave   = tid >> 5;
  const int  lane   = tid & 31;
  const bool hiHalf = lane >= 16;
  const int  j      = wave * 16 + (lane & 15); // absolute H column (C/D N index)
  const int  bg     = blockIdx.x;              // batch group: rows bg*16..bg*16+15
  const int  mrow   = lane & 15;               // A-operand row this lane gathers

  // zero initial hidden state (buffer 0 is read at t=0)
  for (int i = tid; i < 16 * LDS_STRIDE; i += 256) {
    h0s[0][i] = (__bf16)0.f;
    h1s[0][i] = (__bf16)0.f;
  }

  // ---- weight B-tiles into registers once (loop-invariant over T) ----
  const int KBo = hiHalf ? 8 : 0;    // B-fragment K offset for this lane half
  const int KAo = hiHalf ? 8  : 0;   // A-fragment K offset for this lane half
  v16bf Bih0[2], Bhh0[4], Bih1[4], Bhh1[4];
#pragma unroll
  for (int kb = 0; kb < 2; ++kb)
    Bih0[kb] = loadB_f32(Wih0 + (size_t)j * D_IN, kb * 32 + KBo);
#pragma unroll
  for (int kb = 0; kb < 4; ++kb) {
    Bhh0[kb] = loadB_f32(Whh0 + (size_t)j * H_DIM, kb * 32 + KBo);
    Bih1[kb] = loadB_f32(Wih1 + (size_t)j * H_DIM, kb * 32 + KBo);
    Bhh1[kb] = loadB_f32(Whh1 + (size_t)j * H_DIM, kb * 32 + KBo);
  }
  const float bias0 = bih0[j] + bhh0[j];
  const float bias1 = bih1[j] + bhh1[j];

  const float* xrow = x + (size_t)(bg * 16 + mrow) * T_LEN * D_IN;

  // software-pipelined x[t] A-fragments (t = 0 primed here)
  v16bf ax0 = loadA_f32(xrow, KAo);
  v16bf ax1 = loadA_f32(xrow, 32 + KAo);

  __syncthreads();

  for (int t = 0; t < T_LEN; ++t) {
    const int cur = t & 1;
    const __bf16* h0rd  = &h0s[cur][mrow * LDS_STRIDE];      // h0[t-1]
    const __bf16* h0rdN = &h0s[cur ^ 1][mrow * LDS_STRIDE];  // h0[t] (after store)
    const __bf16* h1rd  = &h1s[cur][mrow * LDS_STRIDE];      // h1[t-1]
    __bf16* h0wr = &h0s[cur ^ 1][0];
    __bf16* h1wr = &h1s[cur ^ 1][0];

    // issue next step's x loads now; consumed at t+1 (clamped, branch-free)
    const float* xn = xrow + (size_t)((t + 1 < T_LEN) ? (t + 1) : t) * D_IN;
    v16bf nx0 = loadA_f32(xn, KAo);
    v16bf nx1 = loadA_f32(xn, 32 + KAo);
    __builtin_prefetch(xrow + (size_t)((t + 2 < T_LEN) ? (t + 2) : t) * D_IN, 0, 0);

    // ---- layer 0: c0 = x_t @ Wih0^T + h0[t-1] @ Whh0^T + bias ----
    v8f c0 = { bias0, bias0, bias0, bias0, bias0, bias0, bias0, bias0 };
    c0 = __builtin_amdgcn_wmma_f32_16x16x32_bf16(false, ax0, false, Bih0[0],
                                                 (short)0, c0, false, false);
    c0 = __builtin_amdgcn_wmma_f32_16x16x32_bf16(false, ax1, false, Bih0[1],
                                                 (short)0, c0, false, false);
    v16bf a0[4];
#pragma unroll
    for (int kb = 0; kb < 4; ++kb) a0[kb] = loadA_lds(h0rd, kb * 32 + KAo);
#pragma unroll
    for (int kb = 0; kb < 4; ++kb)
      c0 = __builtin_amdgcn_wmma_f32_16x16x32_bf16(false, a0[kb], false, Bhh0[kb],
                                                   (short)0, c0, false, false);
#pragma unroll
    for (int r = 0; r < 8; ++r) {   // C layout: VGPR r -> M=r (+8 for hi half)
      const int m = hiHalf ? r + 8 : r;
      h0wr[m * LDS_STRIDE + j] = (__bf16)relu1(c0[r]);
    }
    __syncthreads();                // h0[t] visible (ping-pong: no WAR barrier)

    // ---- layer 1: c1 = h0[t] @ Wih1^T + h1[t-1] @ Whh1^T + bias ----
    v8f c1 = { bias1, bias1, bias1, bias1, bias1, bias1, bias1, bias1 };
    v16bf ai[4], ah[4];
#pragma unroll
    for (int kb = 0; kb < 4; ++kb) {
      ai[kb] = loadA_lds(h0rdN, kb * 32 + KAo);
      ah[kb] = loadA_lds(h1rd,  kb * 32 + KAo);
    }
#pragma unroll
    for (int kb = 0; kb < 4; ++kb) {
      c1 = __builtin_amdgcn_wmma_f32_16x16x32_bf16(false, ai[kb], false, Bih1[kb],
                                                   (short)0, c1, false, false);
      c1 = __builtin_amdgcn_wmma_f32_16x16x32_bf16(false, ah[kb], false, Bhh1[kb],
                                                   (short)0, c1, false, false);
    }
#pragma unroll
    for (int r = 0; r < 8; ++r) {
      const int m = hiHalf ? r + 8 : r;
      h1wr[m * LDS_STRIDE + j] = (__bf16)relu1(c1[r]);
    }
    __syncthreads();                // h1[t] visible

    ax0 = nx0;
    ax1 = nx1;
  }

  // output: h1[b, T-1, H-1]; last write went to buffer ((T-1)&1)^1
  if (tid < 16)
    out[bg * 16 + tid] =
        (float)h1s[((T_LEN - 1) & 1) ^ 1][tid * LDS_STRIDE + (H_DIM - 1)];
}

extern "C" void kernel_launch(void* const* d_in, const int* in_sizes, int n_in,
                              void* d_out, int out_size, void* d_ws, size_t ws_size,
                              hipStream_t stream) {
  const float* x    = (const float*)d_in[0];
  const float* Wih0 = (const float*)d_in[1];
  const float* Whh0 = (const float*)d_in[2];
  const float* bih0 = (const float*)d_in[3];
  const float* bhh0 = (const float*)d_in[4];
  const float* Wih1 = (const float*)d_in[5];
  const float* Whh1 = (const float*)d_in[6];
  const float* bih1 = (const float*)d_in[7];
  const float* bhh1 = (const float*)d_in[8];
  float* out = (float*)d_out;

  rnn2_fused<<<dim3(B_TOT / 16), dim3(256), 0, stream>>>(
      x, Wih0, Whh0, bih0, bhh0, Wih1, Whh1, bih1, bhh1, out);

  (void)in_sizes; (void)n_in; (void)out_size; (void)d_ws; (void)ws_size;
}
